// OptimizedGroupedQueryAttention_89386859364713
// MI455X (gfx1250) — matrix operand order repaired and checked by reading, not compiled, not run on hardware
//
#include <hip/hip_runtime.h>

// ---------------------------------------------------------------------------
// GQA attention block for MI455X (gfx1250, wave32, WMMA + TDM).
// [GEMM Wq/Wk/Wv, bf16 WMMA, double-buffered LDS] ->
// [RMSNorm+RoPE relayout to bf16; Q pre-scaled by 0.125*log2(e); V stored
//  pre-transposed] ->
// [flash causal attention: TDM tile loads, exp2-based online softmax,
//  O computed transposed for contiguous b128 stores] -> [GEMM Wo].
// ---------------------------------------------------------------------------

typedef __bf16 v16bf __attribute__((ext_vector_type(16)));
typedef float  v8f   __attribute__((ext_vector_type(8)));
typedef unsigned int v4u __attribute__((ext_vector_type(4)));
typedef int          v4i __attribute__((ext_vector_type(4)));
typedef int          v8i __attribute__((ext_vector_type(8)));

union Frag { v16bf v; uint4 u[2]; };

__device__ __forceinline__ unsigned short to_bf16(float f) {
  union { __bf16 b; unsigned short u; } cv;
  cv.b = (__bf16)f;
  return cv.u;
}

__device__ __forceinline__ v8f v8zero() {
  v8f z;
  #pragma unroll
  for (int i = 0; i < 8; ++i) z[i] = 0.0f;
  return z;
}

__device__ __forceinline__ v8f wmma_bf16(const Frag& a, const Frag& b, v8f c) {
  return __builtin_amdgcn_wmma_f32_16x16x32_bf16(
      false, a.v, false, b.v, (short)0, c, false, false);
}

// ---------------------------------------------------------------------------
// Tensor Data Mover: 2D tile (global -> LDS), bf16 elements, LDS row padding.
// ---------------------------------------------------------------------------
#if __has_builtin(__builtin_amdgcn_tensor_load_to_lds)
#define HAVE_TDM 1
__device__ __forceinline__ void tdm_load_2d(unsigned lds_off, const void* gptr,
                                            unsigned tile_x, unsigned tile_y,
                                            unsigned tensor_x, unsigned tensor_y,
                                            unsigned stride_el,
                                            unsigned pad_int, unsigned pad_amt) {
  unsigned long long ga = (unsigned long long)(uintptr_t)gptr;
  v4u g0;
  g0[0] = 1u;                                              // count=1 (valid user D#)
  g0[1] = lds_off;                                         // lds_addr (bytes)
  g0[2] = (unsigned)ga;                                    // global_addr[31:0]
  g0[3] = ((unsigned)(ga >> 32) & 0x01FFFFFFu) | (2u << 30); // [56:32] | type=2
  unsigned pe = (pad_amt | pad_int) ? 1u : 0u;
  v8i g1;
  g1[0] = (int)((1u << 16) | (pe << 20) | (pad_int << 22) | (pad_amt << 25)); // 2B elems
  g1[1] = (int)((tensor_x & 0xFFFFu) << 16);               // tensor_dim0[15:0]
  g1[2] = (int)((tensor_x >> 16) | ((tensor_y & 0xFFFFu) << 16));
  g1[3] = (int)((tensor_y >> 16) | (tile_x << 16));        // tile_dim0
  g1[4] = (int)tile_y;                                     // tile_dim1 (tile_dim2=0)
  g1[5] = (int)stride_el;                                  // tensor_dim0_stride lo32
  g1[6] = 0;
  g1[7] = 0;
  v4i g2; g2[0] = 1; g2[1] = 1; g2[2] = 0; g2[3] = 0;      // dim2/dim3 = 1
  v4i g3; g3[0] = 0; g3[1] = (int)(1u << 16); g3[2] = 0; g3[3] = 0; // dim4 = 1
#if defined(__clang_major__) && __clang_major__ >= 23
  v8i g4; g4[0]=0; g4[1]=0; g4[2]=0; g4[3]=0; g4[4]=0; g4[5]=0; g4[6]=0; g4[7]=0;
  __builtin_amdgcn_tensor_load_to_lds(g0, g1, g2, g3, g4, 0);
#else
  __builtin_amdgcn_tensor_load_to_lds(g0, g1, g2, g3, 0);
#endif
}
#else
#define HAVE_TDM 0
#endif

#if defined(__HIP_DEVICE_COMPILE__)
#if HAVE_TDM
#warning "gfx1250: TDM path ENABLED (__builtin_amdgcn_tensor_load_to_lds)"
#else
#warning "gfx1250: TDM builtin NOT available -> vector-copy fallback"
#endif
#endif

// ---------------------------------------------------------------------------
// GEMM: C[M,N] = A[M,K](f32) @ B[K,N](f32), bf16 WMMA, f32 accumulate.
// BM=256, BN=128, BK=32; 8 waves (4m x 2n), wave tile 64x64 = 16 WMMA per
// k-step against 16 LDS b128 frag loads. Double-buffered LDS, one barrier
// per k-step; fetch(k+1) issues before the WMMAs of step k.
// ---------------------------------------------------------------------------
#define GP 40

template<int N, int K>
__global__ __launch_bounds__(256)
void gemm_bf16w(const float* __restrict__ A, const float* __restrict__ B,
                float* __restrict__ C)
{
  __shared__ __align__(16) unsigned short As[2][256 * GP];
  __shared__ __align__(16) unsigned short Bs[2][128 * GP];

  const int tid  = threadIdx.x;
  const int lane = tid & 31, wid = tid >> 5;
  const int hl = lane >> 4, l16 = lane & 15;
  const int m0 = blockIdx.y * 256, n0 = blockIdx.x * 128;
  const int wm = (wid >> 1) * 64, wn = (wid & 1) * 64;

  v8f acc[4][4];
  #pragma unroll
  for (int i = 0; i < 4; ++i)
    #pragma unroll
    for (int j = 0; j < 4; ++j) acc[i][j] = v8zero();

  const int bk = (tid & 7) * 4, bn = (tid >> 3) * 4;  // B: 4x4 sub-blocks

  float4 ar[8];      // A: one 256-elem row slice (32 f32) per thread
  float4 br[4];      // B: 4x4 f32 sub-block per thread

  auto fetch = [&](int k0) {
    const float4* asrc =
        reinterpret_cast<const float4*>(A + (size_t)(m0 + tid) * K + k0);
    #pragma unroll
    for (int i = 0; i < 8; ++i) ar[i] = asrc[i];
    #pragma unroll
    for (int kk = 0; kk < 4; ++kk)
      br[kk] = *reinterpret_cast<const float4*>(
          B + (size_t)(k0 + bk + kk) * N + n0 + bn);
  };

  auto stage = [&](int buf) {
    unsigned short h[32] __attribute__((aligned(16)));
    #pragma unroll
    for (int i = 0; i < 8; ++i) {
      h[4*i+0] = to_bf16(ar[i].x); h[4*i+1] = to_bf16(ar[i].y);
      h[4*i+2] = to_bf16(ar[i].z); h[4*i+3] = to_bf16(ar[i].w);
    }
    uint4* dst = reinterpret_cast<uint4*>(&As[buf][tid * GP]);
    const uint4* h4 = reinterpret_cast<const uint4*>(h);
    dst[0] = h4[0]; dst[1] = h4[1]; dst[2] = h4[2]; dst[3] = h4[3];
    unsigned short hb[4][4];
    #pragma unroll
    for (int kk = 0; kk < 4; ++kk) {
      hb[kk][0] = to_bf16(br[kk].x); hb[kk][1] = to_bf16(br[kk].y);
      hb[kk][2] = to_bf16(br[kk].z); hb[kk][3] = to_bf16(br[kk].w);
    }
    #pragma unroll
    for (int nn = 0; nn < 4; ++nn) {
      unsigned short pk[4] __attribute__((aligned(8))) =
          { hb[0][nn], hb[1][nn], hb[2][nn], hb[3][nn] };
      *reinterpret_cast<uint2*>(&Bs[buf][(bn + nn) * GP + bk]) =
          *reinterpret_cast<const uint2*>(pk);
    }
  };

  fetch(0);
  stage(0);
  __syncthreads();

  int buf = 0;
  for (int k0 = 0; k0 < K; k0 += 32) {
    const bool more = (k0 + 32 < K);
    if (more) fetch(k0 + 32);                 // long-latency loads first

    Frag af[4], bfr[4];
    #pragma unroll
    for (int mt = 0; mt < 4; ++mt) {
      int base = (wm + mt*16 + l16) * GP + hl * 8;
      af[mt].u[0] = *reinterpret_cast<const uint4*>(&As[buf][base]);
      af[mt].u[1] = *reinterpret_cast<const uint4*>(&As[buf][base + 16]);
    }
    #pragma unroll
    for (int nt = 0; nt < 4; ++nt) {
      int base = (wn + nt*16 + l16) * GP + hl * 8;
      bfr[nt].u[0] = *reinterpret_cast<const uint4*>(&Bs[buf][base]);
      bfr[nt].u[1] = *reinterpret_cast<const uint4*>(&Bs[buf][base + 16]);
    }
    #pragma unroll
    for (int mt = 0; mt < 4; ++mt)
      #pragma unroll
      for (int nt = 0; nt < 4; ++nt)
        acc[mt][nt] = wmma_bf16(af[mt], bfr[nt], acc[mt][nt]);

    if (more) stage(buf ^ 1);                 // overlap with WMMA issue
    __syncthreads();
    buf ^= 1;
  }

  #pragma unroll
  for (int mt = 0; mt < 4; ++mt)
    #pragma unroll
    for (int nt = 0; nt < 4; ++nt) {
      float* p = C + (size_t)(m0 + wm + mt*16 + hl*8) * N
                   + (n0 + wn + nt*16 + l16);
      #pragma unroll
      for (int r = 0; r < 8; ++r) p[(size_t)r * N] = acc[mt][nt][r];
    }
}

// ---------------------------------------------------------------------------
// RMSNorm + RoPE, relayout to bf16. Q,K head-major [B][h][S][D]; V stored
// PRE-TRANSPOSED [B][kvh][D][S]. Q additionally pre-scaled by
// 0.125*log2(e) so the flash loop needs no scale multiply and can use exp2.
// ---------------------------------------------------------------------------
#define QSCALE 0.18033688011112042f   // (1/sqrt(64)) * log2(e)

__global__ __launch_bounds__(256)
void normrope(const float* __restrict__ qf, const float* __restrict__ kf,
              const float* __restrict__ vf, const float* __restrict__ cosp,
              const float* __restrict__ sinp, const float* __restrict__ qw,
              const float* __restrict__ kw,
              unsigned short* __restrict__ qb, unsigned short* __restrict__ kb,
              unsigned short* __restrict__ vb)
{
  const int lane = threadIdx.x & 31;
  const int wid  = threadIdx.x >> 5;
  const long wg = (long)blockIdx.x * 8 + wid;
  const int  slot  = (int)(wg % 48);
  const long token = wg / 48;                 // 0 .. 4095
  const int b = (int)(token >> 11);
  const int s = (int)(token & 2047);

  if (slot >= 40) {                           // V: convert + transposed store
    int hh = slot - 40;
    const float* src = vf + token * 512 + hh * 64;
    unsigned short* dstv = vb + ((size_t)(b * 8 + hh) * 64) * 2048;
    dstv[(size_t)lane * 2048 + s]        = to_bf16(src[lane]);
    dstv[(size_t)(lane + 32) * 2048 + s] = to_bf16(src[lane + 32]);
    return;
  }

  const float* src; unsigned short* dst; const float* w; float post;
  if (slot < 32) {
    src = qf + token * 2048 + slot * 64;
    dst = qb + (((size_t)(b * 32 + slot)) * 2048 + s) * 64;
    w = qw; post = QSCALE;
  } else {
    int hh = slot - 32;
    src = kf + token * 512 + hh * 64;
    dst = kb + (((size_t)(b * 8 + hh)) * 2048 + s) * 64;
    w = kw; post = 1.0f;
  }

  float x1 = src[lane], x2 = src[lane + 32];
  float ss = x1 * x1 + x2 * x2;
  #pragma unroll
  for (int m = 16; m >= 1; m >>= 1) ss += __shfl_xor(ss, m, 32);
  float rn = rsqrtf(ss * (1.0f / 64.0f) + 1e-6f);
  x1 *= rn * w[lane];
  x2 *= rn * w[lane + 32];
  float c1 = cosp[s * 64 + lane],      s1 = sinp[s * 64 + lane];
  float c2 = cosp[s * 64 + lane + 32], s2 = sinp[s * 64 + lane + 32];
  dst[lane]      = to_bf16((x1 * c1 - x2 * s1) * post);
  dst[lane + 32] = to_bf16((x2 * c2 + x1 * s2) * post);
}

// ---------------------------------------------------------------------------
// Flash-style causal attention (GQA 4:1), bf16 WMMA, exp2-domain softmax
// (Q carries 0.125*log2e). K + V^T tiles via TDM into padded LDS.
// O computed TRANSPOSED: O^T = V^T @ P^T.
// ---------------------------------------------------------------------------
#define KP 72
#define VP 136
#define PP 136

__global__ __launch_bounds__(256)
void attn_fwd(const unsigned short* __restrict__ qb,
              const unsigned short* __restrict__ kb,
              const unsigned short* __restrict__ vb,
              float* __restrict__ out)
{
  __shared__ __align__(16) unsigned short Ks[128 * KP + 8];
  __shared__ __align__(16) unsigned short Vs[64 * VP + 8];
  __shared__ __align__(16) unsigned short Ps[8 * 16 * PP];
  __shared__ float alphaB[8][16];
  __shared__ float lB[8][16];

  const int tid  = threadIdx.x;
  const int lane = tid & 31, wid = tid >> 5;
  const int hl = lane >> 4, l16 = lane & 15;
  const int iq = blockIdx.x, h = blockIdx.y, b = blockIdx.z;
  const int kvh = h >> 2;

  const unsigned short* qptr = qb + (size_t)(b * 32 + h)   * 2048 * 64;
  const unsigned short* kptr = kb + (size_t)(b * 8  + kvh) * 2048 * 64;
  const unsigned short* vptr = vb + (size_t)(b * 8  + kvh) * 64 * 2048; // [D][S]

  const int qr0 = iq * 128 + wid * 16;

  Frag qfrag[2];
  #pragma unroll
  for (int ks = 0; ks < 2; ++ks) {
    size_t off = (size_t)(qr0 + l16) * 64 + ks * 32 + hl * 8;
    qfrag[ks].u[0] = *reinterpret_cast<const uint4*>(qptr + off);
    qfrag[ks].u[1] = *reinterpret_cast<const uint4*>(qptr + off + 16);
  }

  v8f acco[4];                                 // O^T tiles: M=d, N=q-row
  #pragma unroll
  for (int dt = 0; dt < 4; ++dt) acco[dt] = v8zero();
  float mrow[8], lrow[8];
  #pragma unroll
  for (int r = 0; r < 8; ++r) { mrow[r] = -1e30f; lrow[r] = 0.0f; }

  unsigned short* Pw = &Ps[wid * 16 * PP];

  for (int j = 0; j <= iq; ++j) {
    __syncthreads();                           // previous tile fully consumed
#if HAVE_TDM
    if (wid == 0) {
      // K: 128(rows) x 64(d), stride 64; pad 8 halfs/row -> pitch 72
      tdm_load_2d((unsigned)(uintptr_t)&Ks[0], kptr + (size_t)j * 128 * 64,
                  64, 128, 64, 2048, 64, /*pad_int=*/4, /*pad_amt=*/3);
      // V^T: 128(s) x 64(d rows), stride 2048; pad -> pitch 136
      tdm_load_2d((unsigned)(uintptr_t)&Vs[0], vptr + (size_t)j * 128,
                  128, 64, 2048, 64, 2048, /*pad_int=*/5, /*pad_amt=*/3);
      __builtin_amdgcn_s_wait_tensorcnt(0);
    }
#else
    {
      int r = tid >> 1, seg = tid & 1;
      const uint4* ksrc = reinterpret_cast<const uint4*>(
          kptr + (size_t)(j * 128 + r) * 64 + seg * 32);
      uint4* kdst = reinterpret_cast<uint4*>(&Ks[r * KP + seg * 32]);
      kdst[0] = ksrc[0]; kdst[1] = ksrc[1]; kdst[2] = ksrc[2]; kdst[3] = ksrc[3];
      int d = tid >> 2, vseg = tid & 3;
      const uint4* vsrc = reinterpret_cast<const uint4*>(
          vptr + (size_t)d * 2048 + j * 128 + vseg * 32);
      uint4* vdst = reinterpret_cast<uint4*>(&Vs[d * VP + vseg * 32]);
      vdst[0] = vsrc[0]; vdst[1] = vsrc[1]; vdst[2] = vsrc[2]; vdst[3] = vsrc[3];
    }
#endif
    __syncthreads();

    // S = Q @ K^T  (already in exp2 domain, no scale multiply needed)
    v8f sacc[8];
    #pragma unroll
    for (int nt = 0; nt < 8; ++nt) {
      sacc[nt] = v8zero();
      #pragma unroll
      for (int ks = 0; ks < 2; ++ks) {
        Frag kfrag;
        int base = (nt * 16 + l16) * KP + ks * 32 + hl * 8;
        kfrag.u[0] = *reinterpret_cast<const uint4*>(&Ks[base]);
        kfrag.u[1] = *reinterpret_cast<const uint4*>(&Ks[base + 16]);
        sacc[nt] = wmma_bf16(qfrag[ks], kfrag, sacc[nt]);
      }
    }

    // causal mask (diagonal block only)
    if (j == iq) {
      #pragma unroll
      for (int nt = 0; nt < 8; ++nt) {
        int col = j * 128 + nt * 16 + l16;
        #pragma unroll
        for (int r = 0; r < 8; ++r) {
          int row = qr0 + hl * 8 + r;
          if (col > row) sacc[nt][r] = -1e30f;
        }
      }
    }

    // online softmax in exp2 domain (row stats in (vgpr r, half) slots)
    float alpha[8];
    #pragma unroll
    for (int r = 0; r < 8; ++r) {
      float v = sacc[0][r];
      #pragma unroll
      for (int nt = 1; nt < 8; ++nt) v = fmaxf(v, sacc[nt][r]);
      v = fmaxf(v, __shfl_xor(v, 1));
      v = fmaxf(v, __shfl_xor(v, 2));
      v = fmaxf(v, __shfl_xor(v, 4));
      v = fmaxf(v, __shfl_xor(v, 8));
      float mn = fmaxf(mrow[r], v);
      alpha[r] = exp2f(mrow[r] - mn);
      mrow[r] = mn;
      lrow[r] *= alpha[r];
    }
    if (l16 == 0) {
      #pragma unroll
      for (int r = 0; r < 8; ++r) alphaB[wid][hl * 8 + r] = alpha[r];
    }
    float alphaT = alphaB[wid][l16];
    #pragma unroll
    for (int dt = 0; dt < 4; ++dt)
      #pragma unroll
      for (int r = 0; r < 8; ++r) acco[dt][r] *= alphaT;

    // P = exp2(S - m), row sums, stage P (row-major = P^T B-layout)
    float rsum[8];
    #pragma unroll
    for (int r = 0; r < 8; ++r) rsum[r] = 0.0f;
    #pragma unroll
    for (int nt = 0; nt < 8; ++nt)
      #pragma unroll
      for (int r = 0; r < 8; ++r) {
        float p = exp2f(sacc[nt][r] - mrow[r]);
        rsum[r] += p;
        Pw[(hl * 8 + r) * PP + nt * 16 + l16] = to_bf16(p);
      }
    #pragma unroll
    for (int r = 0; r < 8; ++r) {
      float v = rsum[r];
      v += __shfl_xor(v, 1); v += __shfl_xor(v, 2);
      v += __shfl_xor(v, 4); v += __shfl_xor(v, 8);
      lrow[r] += v;
    }

    // O^T += V^T @ P^T
    #pragma unroll
    for (int ks = 0; ks < 4; ++ks) {
      Frag pfrag;
      int pbase = l16 * PP + ks * 32 + hl * 8;
      pfrag.u[0] = *reinterpret_cast<const uint4*>(&Pw[pbase]);
      pfrag.u[1] = *reinterpret_cast<const uint4*>(&Pw[pbase + 16]);
      #pragma unroll
      for (int dt = 0; dt < 4; ++dt) {
        Frag vfrag;
        int vbase = (dt * 16 + l16) * VP + ks * 32 + hl * 8;
        vfrag.u[0] = *reinterpret_cast<const uint4*>(&Vs[vbase]);
        vfrag.u[1] = *reinterpret_cast<const uint4*>(&Vs[vbase + 16]);
        acco[dt] = wmma_bf16(vfrag, pfrag, acco[dt]);
      }
    }
  }

  // normalize + write: O^T element (d, qcol) -> out[b*S+qcol][h*64+d]
  if (l16 == 0) {
    #pragma unroll
    for (int r = 0; r < 8; ++r) lB[wid][hl * 8 + r] = lrow[r];
  }
  __syncthreads();
  float invL = 1.0f / lB[wid][l16];
  #pragma unroll
  for (int dt = 0; dt < 4; ++dt) {
    float* po = out + ((size_t)b * 2048 + qr0 + l16) * 2048
                    + h * 64 + dt * 16 + hl * 8;
    float4 o0 = { acco[dt][0] * invL, acco[dt][1] * invL,
                  acco[dt][2] * invL, acco[dt][3] * invL };
    float4 o1 = { acco[dt][4] * invL, acco[dt][5] * invL,
                  acco[dt][6] * invL, acco[dt][7] * invL };
    reinterpret_cast<float4*>(po)[0] = o0;
    reinterpret_cast<float4*>(po)[1] = o1;
  }
}

// ---------------------------------------------------------------------------
extern "C" void kernel_launch(void* const* d_in, const int* in_sizes, int n_in,
                              void* d_out, int out_size, void* d_ws, size_t ws_size,
                              hipStream_t stream) {
  (void)in_sizes; (void)n_in; (void)out_size; (void)ws_size;
  const float* hs   = (const float*)d_in[0];
  const float* cosp = (const float*)d_in[1];
  const float* sinp = (const float*)d_in[2];
  const float* Wq   = (const float*)d_in[3];
  const float* Wk   = (const float*)d_in[4];
  const float* Wv   = (const float*)d_in[5];
  const float* Wo   = (const float*)d_in[6];
  const float* qw   = (const float*)d_in[7];
  const float* kw   = (const float*)d_in[8];
  // d_in[9] cu_seqlens, d_in[10] max_s: contiguous full-length sequences here.

  char* ws = (char*)d_ws;
  float*          qf   = (float*)(ws + 0);                 // 4096x2048 f32
  float*          kf   = (float*)(ws + 33554432);          // 4096x512  f32
  float*          vf   = (float*)(ws + 41943040);          // 4096x512  f32
  unsigned short* qb2  = (unsigned short*)(ws + 50331648); // [2][32][2048][64] bf16
  unsigned short* kb2  = (unsigned short*)(ws + 67108864); // [2][8][2048][64]  bf16
  unsigned short* vb2  = (unsigned short*)(ws + 71303168); // [2][8][64][2048]  bf16 (V^T)
  float*          attn = (float*)(ws + 75497472);          // 4096x2048 f32
  float*          outp = (float*)d_out;                    // 4096x2048 f32

  dim3 blk(256);
  gemm_bf16w<2048, 2048><<<dim3(16, 16), blk, 0, stream>>>(hs, Wq, qf);
  gemm_bf16w< 512, 2048><<<dim3( 4, 16), blk, 0, stream>>>(hs, Wk, kf);
  gemm_bf16w< 512, 2048><<<dim3( 4, 16), blk, 0, stream>>>(hs, Wv, vf);
  normrope<<<24576, blk, 0, stream>>>(qf, kf, vf, cosp, sinp, qw, kw, qb2, kb2, vb2);
  attn_fwd<<<dim3(16, 32, 2), blk, 0, stream>>>(qb2, kb2, vb2, attn);
  gemm_bf16w<2048, 2048><<<dim3(16, 16), blk, 0, stream>>>(attn, Wo, outp);
}
